// variable_selection_network_34729105555839
// MI455X (gfx1250) — compile-verified, matrix operand order checked by reading, not run on hardware
//
#include <hip/hip_runtime.h>
#include <hip/hip_bf16.h>
#include <stdint.h>

#define B_SZ 2048
#define V_SZ 96
#define D_SZ 128
#define F_SZ (V_SZ * D_SZ) /* 12288 */
#define CD_SZ 2048
#define LN_EPS 1e-3f

typedef __attribute__((ext_vector_type(16))) __bf16 v16bf;
typedef __attribute__((ext_vector_type(2)))  __bf16 v2bf;
typedef __attribute__((ext_vector_type(8)))  float  v8f;

#define WMMA_BF16(acc, a, b)                                                   \
  acc = __builtin_amdgcn_wmma_f32_16x16x32_bf16(false, (a), false, (b),        \
                                                (short)0, (acc), false, false)

struct U32x8 { uint4 lo; uint4 hi; };

__device__ __forceinline__ uint16_t f2bf(float f) {
  uint32_t u = __float_as_uint(f);
  u += 0x7fffu + ((u >> 16) & 1u); // round to nearest even
  return (uint16_t)(u >> 16);
}

__device__ __forceinline__ v8f vzero8() {
  v8f z;
#pragma unroll
  for (int i = 0; i < 8; ++i) z[i] = 0.f;
  return z;
}

// A fragment (16x32 bf16) from a packed-bf16 row (u32 = K-pair), this lane's
// row = M. Layout per CDNA5 ISA: lane = M + 16*h; word j holds
// K = kbase + (j/4)*16 + h*8 + (j%4)*2 (+1). Words j=0..3 and j=4..7 are each
// 16 contiguous bytes -> two b128 loads, no conversion VALU.
__device__ __forceinline__ v16bf load_a_frag_u32(const uint32_t* rowu,
                                                 int kbase, int h) {
  const uint32_t* p = rowu + (kbase >> 1) + (h << 2);
  U32x8 t;
  t.lo = *(const uint4*)p;
  t.hi = *(const uint4*)(p + 8);
  return __builtin_bit_cast(v16bf, t);
}

// Pre-swizzled B fragment: each lane reads 32 contiguous bytes.
__device__ __forceinline__ v16bf load_b_frag(const uint32_t* __restrict__ W,
                                             int kc, int nt, int ntiles,
                                             int lane) {
  const uint32_t* p = W + (((size_t)(kc * ntiles + nt)) << 8) + (lane << 3);
  U32x8 t;
  t.lo = *(const uint4*)p;
  t.hi = *(const uint4*)(p + 4);
  return __builtin_bit_cast(v16bf, t);
}

// ---------------------------------------------------------------------------
// fp32 -> packed bf16 (one u32 K-pair per thread), hardware cvt.
// ---------------------------------------------------------------------------
__global__ void f32_to_bf16pk(const float* __restrict__ in,
                              uint32_t* __restrict__ out, long nWords) {
  long i = (long)blockIdx.x * blockDim.x + threadIdx.x;
  if (i >= nWords) return;
  const float2 f = ((const float2*)in)[i];
  v2bf b;
  b[0] = (__bf16)f.x;
  b[1] = (__bf16)f.y;
  out[i] = __builtin_bit_cast(uint32_t, b);
}

// ---------------------------------------------------------------------------
// Weight pre-swizzle: fp32 (K x N row-major, nMats batched) -> bf16 fragment
// layout. One thread per packed u32 word.
// ---------------------------------------------------------------------------
__global__ void swizzle_w(const float* __restrict__ W, uint32_t* __restrict__ out,
                          int K, int N, long inMatStride, long outMatStride,
                          int nMats) {
  long wordsPerMat = (long)K * N / 2;
  long tid = (long)blockIdx.x * blockDim.x + threadIdx.x;
  if (tid >= wordsPerMat * nMats) return;
  int mat = (int)(tid / wordsPerMat);
  long lin = tid - (long)mat * wordsPerMat;
  int w    = (int)(lin & 7);
  int lane = (int)((lin >> 3) & 31);
  long t   = lin >> 8;
  int ntiles = N >> 4;
  int nt = (int)(t % ntiles);
  int kc = (int)(t / ntiles);
  int h = lane >> 4;
  int n = (nt << 4) + (lane & 15);
  int k0 = (kc << 5) + ((w >> 2) << 4) + (h << 3) + ((w & 3) << 1);
  const float* Wm = W + (long)mat * inMatStride;
  uint32_t lo = f2bf(Wm[(long)k0 * N + n]);
  uint32_t hi = f2bf(Wm[(long)(k0 + 1) * N + n]);
  out[(long)mat * outMatStride + lin] = lo | (hi << 16);
}

// ---------------------------------------------------------------------------
// Gating GEMMs: eta2 = a@f_W2 + c@f_Wc + b2 ; skip = a@f_Ws + bs.
// Block = 2 waves; wave 0 -> eta2 path (with CD tail), wave 1 -> skip path.
// A side comes from packed bf16 (pure b128 loads).
// ---------------------------------------------------------------------------
__global__ void __launch_bounds__(64)
gating_gemm(const uint32_t* __restrict__ xbf, const uint32_t* __restrict__ cbf,
            const uint32_t* __restrict__ w2s, const uint32_t* __restrict__ wcs,
            const uint32_t* __restrict__ wss, const float* __restrict__ f_b2,
            const float* __restrict__ f_bs, float* __restrict__ eta2,
            float* __restrict__ skip) {
  int wave = threadIdx.x >> 5;
  int lane = threadIdx.x & 31;
  int b0 = blockIdx.x << 4;
  int m16 = lane & 15;
  int h = lane >> 4;

  v8f acc[6];
#pragma unroll
  for (int nt = 0; nt < 6; ++nt) acc[nt] = vzero8();

  const uint32_t* xrow = xbf + (size_t)(b0 + m16) * (F_SZ / 2);
  const uint32_t* Wmain = (wave == 0) ? w2s : wss;

  for (int kc = 0; kc < F_SZ / 32; ++kc) {
    v16bf a = load_a_frag_u32(xrow, kc * 32, h);
    // prefetch the B-fragment stream a few K-chunks ahead (global_prefetch_b8)
    __builtin_prefetch(
        (const void*)(Wmain + (((size_t)((kc + 8) * 6)) << 8) + (lane << 3)),
        0, 1);
#pragma unroll
    for (int nt = 0; nt < 6; ++nt)
      WMMA_BF16(acc[nt], a, load_b_frag(Wmain, kc, nt, 6, lane));
  }
  if (wave == 0) {
    const uint32_t* crow = cbf + (size_t)(b0 + m16) * (CD_SZ / 2);
    for (int kc = 0; kc < CD_SZ / 32; ++kc) {
      v16bf a = load_a_frag_u32(crow, kc * 32, h);
#pragma unroll
      for (int nt = 0; nt < 6; ++nt)
        WMMA_BF16(acc[nt], a, load_b_frag(wcs, kc, nt, 6, lane));
    }
  }

  float* dst = (wave == 0) ? eta2 : skip;
  const float* bias = (wave == 0) ? f_b2 : f_bs;
#pragma unroll
  for (int nt = 0; nt < 6; ++nt) {
    int n = (nt << 4) + m16;
#pragma unroll
    for (int r = 0; r < 8; ++r) {
      int mm = r + (h << 3); // C/D layout: vgpr r, lane-half h -> row
      dst[(size_t)(b0 + mm) * V_SZ + n] = acc[nt][r] + bias[n];
    }
  }
}

// ---------------------------------------------------------------------------
// Small V x V chain + LN + softmax -> gate (B x V). One block per batch row.
// ---------------------------------------------------------------------------
__global__ void __launch_bounds__(128)
gating_chain(const float* __restrict__ eta2, const float* __restrict__ skip,
             const float* __restrict__ f_W1, const float* __restrict__ f_b1,
             const float* __restrict__ f_W4, const float* __restrict__ f_b4,
             const float* __restrict__ f_W5, const float* __restrict__ f_b5,
             const float* __restrict__ f_g, const float* __restrict__ f_be,
             float* __restrict__ gate) {
  __shared__ float shA[V_SZ];
  __shared__ float shB[V_SZ];
  __shared__ float red[128];
  int b = blockIdx.x;
  int n = threadIdx.x;

  if (n < V_SZ) {
    float e = eta2[(size_t)b * V_SZ + n];
    shA[n] = (e > 0.f) ? e : (__expf(e) - 1.f); // elu
  }
  __syncthreads();
  if (n < V_SZ) {
    float s = f_b1[n];
    for (int k = 0; k < V_SZ; ++k) s += shA[k] * f_W1[k * V_SZ + n];
    shB[n] = s; // eta1
  }
  __syncthreads();
  float pre = 0.f;
  if (n < V_SZ) {
    float s4 = f_b4[n], s5 = f_b5[n];
    for (int k = 0; k < V_SZ; ++k) {
      float e1 = shB[k];
      s4 += e1 * f_W4[k * V_SZ + n];
      s5 += e1 * f_W5[k * V_SZ + n];
    }
    float glu = (1.f / (1.f + __expf(-s4))) * s5;
    pre = skip[(size_t)b * V_SZ + n] + glu;
  }
  red[n] = (n < V_SZ) ? pre : 0.f;
  __syncthreads();
  for (int s = 64; s > 0; s >>= 1) {
    if (n < s) red[n] += red[n + s];
    __syncthreads();
  }
  float mean = red[0] / (float)V_SZ;
  __syncthreads();
  red[n] = (n < V_SZ) ? (pre - mean) * (pre - mean) : 0.f;
  __syncthreads();
  for (int s = 64; s > 0; s >>= 1) {
    if (n < s) red[n] += red[n + s];
    __syncthreads();
  }
  float var = red[0] / (float)V_SZ;
  float ln = 0.f;
  if (n < V_SZ)
    ln = (pre - mean) * rsqrtf(var + LN_EPS) * f_g[n] + f_be[n];
  __syncthreads();
  red[n] = (n < V_SZ) ? ln : -1e30f;
  __syncthreads();
  for (int s = 64; s > 0; s >>= 1) {
    if (n < s) red[n] = fmaxf(red[n], red[n + s]);
    __syncthreads();
  }
  float mx = red[0];
  float ex = (n < V_SZ) ? __expf(ln - mx) : 0.f;
  __syncthreads();
  red[n] = ex;
  __syncthreads();
  for (int s = 64; s > 0; s >>= 1) {
    if (n < s) red[n] += red[n + s];
    __syncthreads();
  }
  float inv = 1.f / red[0];
  if (n < V_SZ) gate[(size_t)b * V_SZ + n] = ex * inv;
}

__global__ void zero_out(float* __restrict__ out, int nf) {
  int i = blockIdx.x * blockDim.x + threadIdx.x;
  if (i < nf) out[i] = 0.f;
}

// ---------------------------------------------------------------------------
// Fused per-variable pipeline. Wave owns a 16-row batch tile; block = 4 waves;
// grid = (B/64, V/8). Per variable: 4 chained 16x128x128 bf16 WMMA GEMMs.
// LDS (per-wave, no barriers): sBF = bf16 activation tile in row-major so
// A-fragment reads are two ds_load_b128; sP = fp32 pre-LN stash for exact LN.
// ---------------------------------------------------------------------------
__global__ void __launch_bounds__(128)
pv_fused(const float* __restrict__ x, const uint32_t* __restrict__ xbf,
         const uint32_t* __restrict__ pvs, const float* __restrict__ pv_b2,
         const float* __restrict__ pv_b1, const float* __restrict__ pv_b4,
         const float* __restrict__ pv_b5, const float* __restrict__ pv_g,
         const float* __restrict__ pv_be, const float* __restrict__ gate,
         float* __restrict__ out) {
  __shared__ __bf16 scrBF[4][16][D_SZ]; // 16 KB: A-source relayout
  __shared__ float scrP[4][16][D_SZ];   // 32 KB: fp32 pre-LN stash
  int wave = threadIdx.x >> 5;
  int lane = threadIdx.x & 31;
  int m16 = lane & 15;
  int h = lane >> 4;
  int b0 = (blockIdx.x * 4 + wave) * 16;
  __bf16(*sBF)[D_SZ] = scrBF[wave];
  float(*sP)[D_SZ] = scrP[wave];

  float oacc[8][8];
#pragma unroll
  for (int i = 0; i < 8; ++i)
#pragma unroll
    for (int j = 0; j < 8; ++j) oacc[i][j] = 0.f;

  int v0 = blockIdx.y * 8;
  for (int vi = 0; vi < 8; ++vi) {
    int v = v0 + vi;
    const uint32_t* W2s = pvs + ((size_t)(0 * V_SZ + v) << 13);
    const uint32_t* W1s = pvs + ((size_t)(1 * V_SZ + v) << 13);
    const uint32_t* W4s = pvs + ((size_t)(2 * V_SZ + v) << 13);
    const uint32_t* W5s = pvs + ((size_t)(3 * V_SZ + v) << 13);
    const float* b2 = pv_b2 + (size_t)v * D_SZ;
    const float* b1 = pv_b1 + (size_t)v * D_SZ;
    const float* b4 = pv_b4 + (size_t)v * D_SZ;
    const float* b5 = pv_b5 + (size_t)v * D_SZ;
    const float* gg = pv_g + (size_t)v * D_SZ;
    const float* gb = pv_be + (size_t)v * D_SZ;

    // ---- GEMM1: e2 = xs @ W2 (A direct from packed bf16 x) ----
    v8f acc[8];
#pragma unroll
    for (int nt = 0; nt < 8; ++nt) acc[nt] = vzero8();
    const uint32_t* xrowu =
        xbf + (((size_t)(b0 + m16) * V_SZ + v) * D_SZ >> 1);
#pragma unroll
    for (int kc = 0; kc < 4; ++kc) {
      v16bf a = load_a_frag_u32(xrowu, kc * 32, h);
#pragma unroll
      for (int nt = 0; nt < 8; ++nt)
        WMMA_BF16(acc[nt], a, load_b_frag(W2s, kc, nt, 8, lane));
    }
    // bias + elu -> bf16 LDS (A-layout source for GEMM2)
#pragma unroll
    for (int nt = 0; nt < 8; ++nt) {
      int n = (nt << 4) + m16;
#pragma unroll
      for (int r = 0; r < 8; ++r) {
        int mm = r + (h << 3);
        float e = acc[nt][r] + b2[n];
        sBF[mm][n] = (__bf16)((e > 0.f) ? e : (__expf(e) - 1.f));
      }
    }

    // ---- GEMM2: e1 = elu(e2) @ W1 + b1 -> overwrite bf16 LDS ----
#pragma unroll
    for (int nt = 0; nt < 8; ++nt) acc[nt] = vzero8();
#pragma unroll
    for (int kc = 0; kc < 4; ++kc) {
      v16bf a =
          load_a_frag_u32((const uint32_t*)&sBF[m16][0], kc * 32, h);
#pragma unroll
      for (int nt = 0; nt < 8; ++nt)
        WMMA_BF16(acc[nt], a, load_b_frag(W1s, kc, nt, 8, lane));
    }
#pragma unroll
    for (int nt = 0; nt < 8; ++nt) {
      int n = (nt << 4) + m16;
#pragma unroll
      for (int r = 0; r < 8; ++r)
        sBF[r + (h << 3)][n] = (__bf16)(acc[nt][r] + b1[n]);
    }

    // ---- GEMM3+4: t4 = e1@W4, t5 = e1@W5 (shared A fragments) ----
    v8f acc5[8];
#pragma unroll
    for (int nt = 0; nt < 8; ++nt) { acc[nt] = vzero8(); acc5[nt] = vzero8(); }
#pragma unroll
    for (int kc = 0; kc < 4; ++kc) {
      v16bf a =
          load_a_frag_u32((const uint32_t*)&sBF[m16][0], kc * 32, h);
#pragma unroll
      for (int nt = 0; nt < 8; ++nt) {
        WMMA_BF16(acc[nt], a, load_b_frag(W4s, kc, nt, 8, lane));
        WMMA_BF16(acc5[nt], a, load_b_frag(W5s, kc, nt, 8, lane));
      }
    }

    // ---- GLU + residual (fp32 x); stash pre-LN; accumulate LN stats ----
    float s1[8], s2[8];
#pragma unroll
    for (int r = 0; r < 8; ++r) { s1[r] = 0.f; s2[r] = 0.f; }
#pragma unroll
    for (int nt = 0; nt < 8; ++nt) {
      int n = (nt << 4) + m16;
#pragma unroll
      for (int r = 0; r < 8; ++r) {
        int mm = r + (h << 3);
        float t4 = acc[nt][r] + b4[n];
        float t5 = acc5[nt][r] + b5[n];
        float g2 = (1.f / (1.f + __expf(-t4))) * t5;
        float xs = x[((size_t)(b0 + mm) * V_SZ + v) * D_SZ + n];
        float p = xs + g2;
        sP[mm][n] = p;
        s1[r] += p;
        s2[r] += p * p;
      }
    }
    // Row m = r + 8*h lives in one 16-lane half; xor <= 8 stays in-half.
#pragma unroll
    for (int r = 0; r < 8; ++r) {
#pragma unroll
      for (int off = 1; off <= 8; off <<= 1) {
        s1[r] += __shfl_xor(s1[r], off, 32);
        s2[r] += __shfl_xor(s2[r], off, 32);
      }
    }
    float gv[8], mean[8], rstd[8];
#pragma unroll
    for (int r = 0; r < 8; ++r) {
      mean[r] = s1[r] * (1.f / (float)D_SZ);
      float var = s2[r] * (1.f / (float)D_SZ) - mean[r] * mean[r];
      rstd[r] = rsqrtf(var + LN_EPS);
      gv[r] = gate[(size_t)(b0 + r + (h << 3)) * V_SZ + v];
    }
    // ---- LN + gate-weighted accumulation ----
#pragma unroll
    for (int nt = 0; nt < 8; ++nt) {
      int n = (nt << 4) + m16;
#pragma unroll
      for (int r = 0; r < 8; ++r) {
        int mm = r + (h << 3);
        float hn = (sP[mm][n] - mean[r]) * rstd[r] * gg[n] + gb[n];
        oacc[nt][r] += hn * gv[r];
      }
    }
  }

  // ---- reduce over v-chunks via fp32 atomics into pre-zeroed out ----
#pragma unroll
  for (int nt = 0; nt < 8; ++nt) {
    int n = (nt << 4) + m16;
#pragma unroll
    for (int r = 0; r < 8; ++r) {
      int mm = r + (h << 3);
      atomicAdd(&out[(size_t)(b0 + mm) * D_SZ + n], oacc[nt][r]);
    }
  }
}

// ---------------------------------------------------------------------------
extern "C" void kernel_launch(void* const* d_in, const int* in_sizes, int n_in,
                              void* d_out, int out_size, void* d_ws,
                              size_t ws_size, hipStream_t stream) {
  const float* x     = (const float*)d_in[0];
  const float* c_s   = (const float*)d_in[1];
  const float* pv_W2 = (const float*)d_in[2];
  const float* pv_b2 = (const float*)d_in[3];
  const float* pv_W1 = (const float*)d_in[4];
  const float* pv_b1 = (const float*)d_in[5];
  const float* pv_W4 = (const float*)d_in[6];
  const float* pv_b4 = (const float*)d_in[7];
  const float* pv_W5 = (const float*)d_in[8];
  const float* pv_b5 = (const float*)d_in[9];
  const float* pv_g  = (const float*)d_in[10];
  const float* pv_be = (const float*)d_in[11];
  const float* f_W2  = (const float*)d_in[12];
  const float* f_b2  = (const float*)d_in[13];
  const float* f_Wc  = (const float*)d_in[14];
  const float* f_W1  = (const float*)d_in[15];
  const float* f_b1  = (const float*)d_in[16];
  const float* f_W4  = (const float*)d_in[17];
  const float* f_b4  = (const float*)d_in[18];
  const float* f_W5  = (const float*)d_in[19];
  const float* f_b5  = (const float*)d_in[20];
  const float* f_Ws  = (const float*)d_in[21];
  const float* f_bs  = (const float*)d_in[22];
  const float* f_g   = (const float*)d_in[23];
  const float* f_be  = (const float*)d_in[24];

  // workspace layout (u32 units)
  uint32_t* xbf = (uint32_t*)d_ws;            // B*F/2      = 12582912
  uint32_t* cbf = xbf + 12582912;             // B*CD/2     = 2097152
  uint32_t* w2s = cbf + 2097152;              // F*V/2      = 589824
  uint32_t* wss = w2s + 589824;               // F*V/2      = 589824
  uint32_t* wcs = wss + 589824;               // CD*V/2     = 98304
  uint32_t* pvs = wcs + 98304;                // 4*96*8192  = 3145728
  float* eta2 = (float*)(pvs + 3145728);      // B*V
  float* skip = eta2 + (size_t)B_SZ * V_SZ;   // B*V
  float* gate = skip + (size_t)B_SZ * V_SZ;   // B*V

  // 1) activations -> packed bf16
  {
    long words = (long)B_SZ * F_SZ / 2;
    f32_to_bf16pk<<<(int)((words + 255) / 256), 256, 0, stream>>>(x, xbf,
                                                                  words);
  }
  {
    long words = (long)B_SZ * CD_SZ / 2;
    f32_to_bf16pk<<<(int)((words + 255) / 256), 256, 0, stream>>>(c_s, cbf,
                                                                  words);
  }

  // 2) pre-swizzle all weight matrices into bf16 WMMA fragment layout
  {
    long words = (long)F_SZ * V_SZ / 2;
    int blocks = (int)((words + 255) / 256);
    swizzle_w<<<blocks, 256, 0, stream>>>(f_W2, w2s, F_SZ, V_SZ, 0, 0, 1);
    swizzle_w<<<blocks, 256, 0, stream>>>(f_Ws, wss, F_SZ, V_SZ, 0, 0, 1);
  }
  {
    long words = (long)CD_SZ * V_SZ / 2;
    int blocks = (int)((words + 255) / 256);
    swizzle_w<<<blocks, 256, 0, stream>>>(f_Wc, wcs, CD_SZ, V_SZ, 0, 0, 1);
  }
  {
    long words = 96L * 8192;
    int blocks = (int)((words + 255) / 256);
    swizzle_w<<<blocks, 256, 0, stream>>>(pv_W2, pvs + 0L * 96 * 8192, D_SZ,
                                          D_SZ, 16384, 8192, 96);
    swizzle_w<<<blocks, 256, 0, stream>>>(pv_W1, pvs + 1L * 96 * 8192, D_SZ,
                                          D_SZ, 16384, 8192, 96);
    swizzle_w<<<blocks, 256, 0, stream>>>(pv_W4, pvs + 2L * 96 * 8192, D_SZ,
                                          D_SZ, 16384, 8192, 96);
    swizzle_w<<<blocks, 256, 0, stream>>>(pv_W5, pvs + 3L * 96 * 8192, D_SZ,
                                          D_SZ, 16384, 8192, 96);
  }

  // 3) gating GEMMs (WMMA), then scalar chain + LN + softmax
  gating_gemm<<<B_SZ / 16, 64, 0, stream>>>(xbf, cbf, w2s, wcs, wss, f_b2,
                                            f_bs, eta2, skip);
  gating_chain<<<B_SZ, 128, 0, stream>>>(eta2, skip, f_W1, f_b1, f_W4, f_b4,
                                         f_W5, f_b5, f_g, f_be, gate);

  // 4) zero output, then fused per-variable WMMA pipeline with atomic reduce
  zero_out<<<(B_SZ * D_SZ + 255) / 256, 256, 0, stream>>>((float*)d_out,
                                                          B_SZ * D_SZ);
  pv_fused<<<dim3(B_SZ / 64, V_SZ / 8), 128, 0, stream>>>(
      x, xbf, pvs, pv_b2, pv_b1, pv_b4, pv_b5, pv_g, pv_be, gate,
      (float*)d_out);
}
